// MolFormer_21337397527099
// MI455X (gfx1250) — compile-verified, matrix operand order checked by reading, not run on hardware
//
#include <hip/hip_runtime.h>

// ---------------- types ----------------
typedef __bf16 bf16;
typedef unsigned int u32x4 __attribute__((ext_vector_type(4)));
typedef float        v8f   __attribute__((ext_vector_type(8)));
typedef bf16         v16bf __attribute__((ext_vector_type(16)));

union FragU { u32x4 u[2]; v16bf v; };

// CDNA5 async memory->LDS path (gated: falls back to reg staging if absent)
#if defined(__AMDGCN__) && __has_builtin(__builtin_amdgcn_global_load_async_to_lds_b128) && __has_builtin(__builtin_amdgcn_s_wait_asynccnt)
#define USE_ASYNC_LDS 1
typedef int v4i_gcc __attribute__((vector_size(4 * sizeof(int))));
typedef __attribute__((address_space(1))) v4i_gcc* as1_v4i_ptr;
typedef __attribute__((address_space(3))) v4i_gcc* as3_v4i_ptr;
#else
#define USE_ASYNC_LDS 0
#endif

// ---------------- model constants ----------------
constexpr int cB = 16, cS = 2048, cD = 768, cH = 12, cHD = 64, cNF = 32, cNL = 12;
constexpr int cM = cB * cS;          // 32768 rows
constexpr int cKT = cD / 32;         // 24 k-tiles of 32

// =====================================================================
// Weight convert + transpose: w[l][k][n] (f32) -> wt[l][n][k] (bf16)
// grid (24,24,NL), block (32,8)
// =====================================================================
__global__ void wconv_kernel(const float* __restrict__ w, bf16* __restrict__ wt)
{
    __shared__ float tile[32][33];
    const int l = blockIdx.z;
    const float* wl = w + (size_t)l * cD * cD;
    bf16* wtl = wt + (size_t)l * cD * cD;
    const int n0 = blockIdx.x * 32, k0 = blockIdx.y * 32;
    #pragma unroll
    for (int i = 0; i < 4; ++i) {
        int k = k0 + threadIdx.y + i * 8;
        tile[threadIdx.y + i * 8][threadIdx.x] = wl[(size_t)k * cD + n0 + threadIdx.x];
    }
    __syncthreads();
    #pragma unroll
    for (int i = 0; i < 4; ++i) {
        int n = n0 + threadIdx.y + i * 8;
        wtl[(size_t)n * cD + k0 + threadIdx.x] = (bf16)tile[threadIdx.x][threadIdx.y + i * 8];
    }
}

// =====================================================================
// Embedding gather: x[m,:] = emb[idx[m],:]  (f32 + bf16 copies)
// grid cM, block 256
// =====================================================================
__global__ void embed_kernel(const int* __restrict__ idx, const float* __restrict__ emb,
                             float* __restrict__ xf, bf16* __restrict__ xb)
{
    const int m = blockIdx.x, t = threadIdx.x;
    const int tok = idx[m];
    const float* src = emb + (size_t)tok * cD;
    #pragma unroll
    for (int i = 0; i < 3; ++i) {
        int c = t + i * 256;
        float v = src[c];
        xf[(size_t)m * cD + c] = v;
        xb[(size_t)m * cD + c] = (bf16)v;
    }
}

// =====================================================================
// bf16 WMMA GEMM:  out[M=32768, N=768] = A[M,K=768] @ Wt[N,K]^T + bias
// Epilogue: optional residual add, exact GELU, f32 and/or bf16 stores.
// grid (N/128=6, M/128=256), block 256 (8 waves, 4x2 wave grid)
// Staging: async global->LDS (ASYNCcnt) when available, else reg-staged.
// =====================================================================
template<bool GELU, bool RESID, bool F32OUT, bool BF16OUT>
__global__ __launch_bounds__(256) void gemm_bf16_kernel(
    const bf16* __restrict__ A, const bf16* __restrict__ Wt,
    const float* __restrict__ bias, const float* __restrict__ resid,
    float* __restrict__ outF, bf16* __restrict__ outB)
{
    __shared__ __align__(16) bf16 sA[2][128 * 32];
    __shared__ __align__(16) bf16 sB[2][128 * 32];

    const int tid = threadIdx.x;
    const int blockN0 = blockIdx.x * 128;
    const int blockM0 = blockIdx.y * 128;

    const int lane = tid & 31;
    const int w    = tid >> 5;
    const int wm   = (w & 3) * 32;   // wave row offset in block tile
    const int wn   = (w >> 2) * 64;  // wave col offset in block tile
    const int lr   = lane & 15;
    const int kh   = (lane >> 4) * 8;   // K half select per ISA bf16 layout
    const int hi8  = (lane >> 4) * 8;   // D-matrix row offset per lane group

    v8f acc[2][4] = {};

#if USE_ASYNC_LDS
    // ---- async staging: memory -> LDS without VGPR round trip ----
    auto issueTiles = [&](int kt, int buf) {
        const int kk = kt * 32;
        #pragma unroll
        for (int i = 0; i < 2; ++i) {
            int c = tid + i * 256;
            int row = c >> 2, offe = (c & 3) * 8;
            const bf16* ga = A  + (size_t)(blockM0 + row) * cD + kk + offe;
            const bf16* gb = Wt + (size_t)(blockN0 + row) * cD + kk + offe;
            bf16* la = &sA[buf][row * 32 + offe];
            bf16* lb = &sB[buf][row * 32 + offe];
            __builtin_amdgcn_global_load_async_to_lds_b128(
                (as1_v4i_ptr)(uintptr_t)ga, (as3_v4i_ptr)(unsigned)(uintptr_t)la, 0, 0);
            __builtin_amdgcn_global_load_async_to_lds_b128(
                (as1_v4i_ptr)(uintptr_t)gb, (as3_v4i_ptr)(unsigned)(uintptr_t)lb, 0, 0);
        }
    };
    issueTiles(0, 0);
#else
    // ---- register-staged fallback ----
    u32x4 ra[2], rb[2];
    auto loadTiles = [&](int kt) {
        const int kk = kt * 32;
        #pragma unroll
        for (int i = 0; i < 2; ++i) {
            int c = tid + i * 256;
            int row = c >> 2, offe = (c & 3) * 8;
            ra[i] = *reinterpret_cast<const u32x4*>(A  + (size_t)(blockM0 + row) * cD + kk + offe);
            rb[i] = *reinterpret_cast<const u32x4*>(Wt + (size_t)(blockN0 + row) * cD + kk + offe);
        }
    };
    auto storeTiles = [&](int buf) {
        #pragma unroll
        for (int i = 0; i < 2; ++i) {
            int c = tid + i * 256;
            int row = c >> 2, offe = (c & 3) * 8;
            *reinterpret_cast<u32x4*>(&sA[buf][row * 32 + offe]) = ra[i];
            *reinterpret_cast<u32x4*>(&sB[buf][row * 32 + offe]) = rb[i];
        }
    };
    loadTiles(0);
    storeTiles(0);
#endif

    for (int kt = 0; kt < cKT; ++kt) {
#if USE_ASYNC_LDS
        __builtin_amdgcn_s_wait_asynccnt(0);   // this wave's tile-kt loads done
        __syncthreads();                       // everyone's loads done + prev reads closed
        if (kt + 1 < cKT) issueTiles(kt + 1, (kt + 1) & 1);
#else
        __syncthreads();
        if (kt + 1 < cKT) loadTiles(kt + 1);
#endif

        const int buf = kt & 1;
        FragU af[2], bfr[4];
        #pragma unroll
        for (int r = 0; r < 2; ++r) {
            const bf16* p = &sA[buf][(wm + r * 16 + lr) * 32 + kh];
            af[r].u[0] = *reinterpret_cast<const u32x4*>(p);
            af[r].u[1] = *reinterpret_cast<const u32x4*>(p + 16);
        }
        #pragma unroll
        for (int c = 0; c < 4; ++c) {
            const bf16* p = &sB[buf][(wn + c * 16 + lr) * 32 + kh];
            bfr[c].u[0] = *reinterpret_cast<const u32x4*>(p);
            bfr[c].u[1] = *reinterpret_cast<const u32x4*>(p + 16);
        }
        #pragma unroll
        for (int r = 0; r < 2; ++r)
            #pragma unroll
            for (int c = 0; c < 4; ++c)
                acc[r][c] = __builtin_amdgcn_wmma_f32_16x16x32_bf16(
                    false, af[r].v, false, bfr[c].v, (short)0, acc[r][c], false, false);

#if !USE_ASYNC_LDS
        if (kt + 1 < cKT) storeTiles((kt + 1) & 1);
#endif
    }

    // epilogue
    #pragma unroll
    for (int r = 0; r < 2; ++r) {
        #pragma unroll
        for (int c = 0; c < 4; ++c) {
            const int n = blockN0 + wn + c * 16 + lr;
            const float bv = bias[n];
            const int mbase = blockM0 + wm + r * 16 + hi8;
            #pragma unroll
            for (int i = 0; i < 8; ++i) {
                const int m = mbase + i;
                float val = acc[r][c][i] + bv;
                if (RESID) val += resid[(size_t)m * cD + n];
                if (GELU)  val = 0.5f * val * (1.0f + erff(val * 0.70710678118654752f));
                if (F32OUT)  outF[(size_t)m * cD + n] = val;
                if (BF16OUT) outB[(size_t)m * cD + n] = (bf16)val;
            }
        }
    }
}

// =====================================================================
// RoPE + random-feature map: out[bs, h*32+f] = relu(rope(q)[h,:] @ omega[:,f]) (* mask)
// grid cM, block 128
// =====================================================================
__global__ void rope_phi_kernel(const float* __restrict__ qk, const float* __restrict__ omega,
                                const int* __restrict__ mask, float* __restrict__ out, int useMask)
{
    __shared__ float sRow[cD];
    __shared__ float sOm[cHD * cNF];
    const int bs = blockIdx.x, t = threadIdx.x;
    const int s = bs & (cS - 1);

    for (int i = t; i < cHD * cNF; i += 128) sOm[i] = omega[i];

    const float* row = qk + (size_t)bs * cD;
    for (int dd = t; dd < cD; dd += 128) {
        const int d = dd & 63;
        const int j = (d < 32) ? d : d - 32;
        const float inv = __powf(10000.0f, -(float)j * (1.0f / 32.0f));
        const float ang = (float)s * inv;
        float sn, cs;
        __sincosf(ang, &sn, &cs);
        const float x = row[dd];
        const float rot = (d < 32) ? -row[dd + 32] : row[dd - 32];
        sRow[dd] = x * cs + rot * sn;
    }
    __syncthreads();

    float mval = 1.0f;
    if (useMask) mval = (float)mask[bs];

    for (int o = t; o < cH * cNF; o += 128) {
        const int h = o >> 5, f = o & 31;
        const float* qh = &sRow[h * cHD];
        float acc = 0.0f;
        #pragma unroll 8
        for (int d = 0; d < cHD; ++d) acc += qh[d] * sOm[d * cNF + f];
        acc = fmaxf(acc, 0.0f) * mval;
        out[(size_t)bs * (cH * cNF) + o] = acc;
    }
}

// =====================================================================
// KV[b,h,f,d] = sum_s Kf[b,s,h,f] * v[b,s,h,d];  Ks[b,h,f] = sum_s Kf
// Deterministic (no atomics): one block per (b,h), full S loop, LDS staged.
// grid cB*cH, block 256
// =====================================================================
__global__ void kv_accum_kernel(const float* __restrict__ Kf, const float* __restrict__ v,
                                float* __restrict__ KV, float* __restrict__ Ks)
{
    __shared__ float sV[16 * cHD];
    __shared__ float sK[16 * cNF];
    const int bh = blockIdx.x;
    const int b = bh / cH, h = bh % cH;
    const int t = threadIdx.x;
    const int f = t >> 3, db = (t & 7) * 8;

    float acc[8] = {0.f, 0.f, 0.f, 0.f, 0.f, 0.f, 0.f, 0.f};
    float ks = 0.0f;

    for (int s0 = 0; s0 < cS; s0 += 16) {
        __syncthreads();
        for (int i = t; i < 16 * cHD; i += 256) {
            int si = i >> 6, d = i & 63;
            sV[i] = v[(size_t)(b * cS + s0 + si) * cD + h * cHD + d];
        }
        for (int i = t; i < 16 * cNF; i += 256) {
            int si = i >> 5, ff = i & 31;
            sK[i] = Kf[(size_t)(b * cS + s0 + si) * (cH * cNF) + h * cNF + ff];
        }
        __syncthreads();
        #pragma unroll
        for (int si = 0; si < 16; ++si) {
            const float kf = sK[si * cNF + f];
            #pragma unroll
            for (int j = 0; j < 8; ++j) acc[j] += kf * sV[si * cHD + db + j];
            if ((t & 7) == 0) ks += kf;
        }
    }
    float* kvp = KV + (size_t)bh * (cNF * cHD) + f * cHD + db;
    #pragma unroll
    for (int j = 0; j < 8; ++j) kvp[j] = acc[j];
    if ((t & 7) == 0) Ks[bh * cNF + f] = ks;
}

// =====================================================================
// a[b,s,h,d] = Z(b,s,h) * sum_f Qf[b,s,h,f] * KV[b,h,f,d]  -> bf16 [M,D]
// grid (S/32, cB*cH), block 256
// =====================================================================
__global__ void attn_combine_kernel(const float* __restrict__ Qf, const float* __restrict__ KV,
                                    const float* __restrict__ Ks, bf16* __restrict__ ab)
{
    __shared__ float sKV[cNF * cHD];
    __shared__ float sKs[cNF];
    __shared__ float sQ[32 * cNF];
    __shared__ float sZ[32];
    const int bh = blockIdx.y;
    const int b = bh / cH, h = bh % cH;
    const int s0 = blockIdx.x * 32;
    const int t = threadIdx.x;

    for (int i = t; i < cNF * cHD; i += 256) sKV[i] = KV[(size_t)bh * (cNF * cHD) + i];
    if (t < cNF) sKs[t] = Ks[bh * cNF + t];
    for (int i = t; i < 32 * cNF; i += 256) {
        int sl = i >> 5, f = i & 31;
        sQ[i] = Qf[(size_t)(b * cS + s0 + sl) * (cH * cNF) + h * cNF + f];
    }
    __syncthreads();
    if (t < 32) {
        float d = 0.0f;
        #pragma unroll
        for (int f = 0; f < cNF; ++f) d += sQ[t * cNF + f] * sKs[f];
        sZ[t] = 1.0f / (d + 1e-6f);
    }
    __syncthreads();

    const int sl = t & 31, db = (t >> 5) * 8;
    float acc[8] = {0.f, 0.f, 0.f, 0.f, 0.f, 0.f, 0.f, 0.f};
    #pragma unroll
    for (int f = 0; f < cNF; ++f) {
        const float qv = sQ[sl * cNF + f];
        #pragma unroll
        for (int j = 0; j < 8; ++j) acc[j] += qv * sKV[f * cHD + db + j];
    }
    const float z = sZ[sl];
    bf16* op = ab + (size_t)(b * cS + s0 + sl) * cD + h * cHD + db;
    #pragma unroll
    for (int j = 0; j < 8; ++j) op[j] = (bf16)(acc[j] * z);
}

// =====================================================================
// LayerNorm over D=768; writes f32 (and optional bf16 copy)
// grid cM, block 256
// =====================================================================
__global__ void ln_kernel(const float* __restrict__ in, const float* __restrict__ g,
                          const float* __restrict__ bta, float* __restrict__ outF,
                          bf16* __restrict__ outB)
{
    __shared__ float red[256];
    const int m = blockIdx.x, t = threadIdx.x;
    const float* row = in + (size_t)m * cD;
    float v0 = row[t], v1 = row[t + 256], v2 = row[t + 512];

    red[t] = v0 + v1 + v2;
    __syncthreads();
    for (int o = 128; o > 0; o >>= 1) { if (t < o) red[t] += red[t + o]; __syncthreads(); }
    const float mean = red[0] * (1.0f / 768.0f);
    __syncthreads();

    const float d0 = v0 - mean, d1 = v1 - mean, d2 = v2 - mean;
    red[t] = d0 * d0 + d1 * d1 + d2 * d2;
    __syncthreads();
    for (int o = 128; o > 0; o >>= 1) { if (t < o) red[t] += red[t + o]; __syncthreads(); }
    const float var = red[0] * (1.0f / 768.0f);
    const float rinv = rsqrtf(var + 1e-5f);

    const float y0 = d0 * rinv * g[t] + bta[t];
    const float y1 = d1 * rinv * g[t + 256] + bta[t + 256];
    const float y2 = d2 * rinv * g[t + 512] + bta[t + 512];
    outF[(size_t)m * cD + t]       = y0;
    outF[(size_t)m * cD + t + 256] = y1;
    outF[(size_t)m * cD + t + 512] = y2;
    if (outB) {
        outB[(size_t)m * cD + t]       = (bf16)y0;
        outB[(size_t)m * cD + t + 256] = (bf16)y1;
        outB[(size_t)m * cD + t + 512] = (bf16)y2;
    }
}

// =====================================================================
// Host driver
// =====================================================================
extern "C" void kernel_launch(void* const* d_in, const int* in_sizes, int n_in,
                              void* d_out, int out_size, void* d_ws, size_t ws_size,
                              hipStream_t stream)
{
    const int*   idx   = (const int*)d_in[0];
    const int*   mask  = (const int*)d_in[1];
    const float* emb   = (const float*)d_in[2];
    const float* wq    = (const float*)d_in[3];
    const float* bq    = (const float*)d_in[4];
    const float* wk    = (const float*)d_in[5];
    const float* bk    = (const float*)d_in[6];
    const float* wv    = (const float*)d_in[7];
    const float* bv    = (const float*)d_in[8];
    const float* wo    = (const float*)d_in[9];
    const float* bo    = (const float*)d_in[10];
    const float* omega = (const float*)d_in[11];
    const float* g1    = (const float*)d_in[12];
    const float* be1   = (const float*)d_in[13];
    const float* w1    = (const float*)d_in[14];
    const float* c1    = (const float*)d_in[15];
    const float* w2    = (const float*)d_in[16];
    const float* c2    = (const float*)d_in[17];
    const float* g2    = (const float*)d_in[18];
    const float* be2   = (const float*)d_in[19];
    const float* gf    = (const float*)d_in[20];
    const float* bfp   = (const float*)d_in[21];

    char* ws = (char*)d_ws;
    size_t off = 0;
    auto alloc = [&](size_t bytes) -> char* {
        char* p = ws + off;
        off += (bytes + 255) & ~(size_t)255;
        return p;
    };
    constexpr size_t DD = (size_t)cD * cD;

    bf16*  wt_all = (bf16*) alloc((size_t)6 * cNL * DD * sizeof(bf16));
    float* xf     = (float*)alloc((size_t)cM * cD * sizeof(float));
    bf16*  xb     = (bf16*) alloc((size_t)cM * cD * sizeof(bf16));
    float* qb     = (float*)alloc((size_t)cM * cD * sizeof(float));   // also residual tmp
    float* kb     = (float*)alloc((size_t)cM * cD * sizeof(float));
    float* vb     = (float*)alloc((size_t)cM * cD * sizeof(float));
    float* Qf     = (float*)alloc((size_t)cM * cH * cNF * sizeof(float));
    float* Kf     = (float*)alloc((size_t)cM * cH * cNF * sizeof(float));
    float* KV     = (float*)alloc((size_t)cB * cH * cNF * cHD * sizeof(float));
    float* Ks     = (float*)alloc((size_t)cB * cH * cNF * sizeof(float));
    bf16*  ab     = (bf16*) alloc((size_t)cM * cD * sizeof(bf16));    // attn out / gelu hidden

    bf16* wtq = wt_all + (size_t)0 * cNL * DD;
    bf16* wtk = wt_all + (size_t)1 * cNL * DD;
    bf16* wtv = wt_all + (size_t)2 * cNL * DD;
    bf16* wto = wt_all + (size_t)3 * cNL * DD;
    bf16* wt1 = wt_all + (size_t)4 * cNL * DD;
    bf16* wt2 = wt_all + (size_t)5 * cNL * DD;

    // ---- one-time (per launch) weight convert/transpose to bf16 [N][K] ----
    {
        dim3 g(24, 24, cNL), b(32, 8);
        wconv_kernel<<<g, b, 0, stream>>>(wq, wtq);
        wconv_kernel<<<g, b, 0, stream>>>(wk, wtk);
        wconv_kernel<<<g, b, 0, stream>>>(wv, wtv);
        wconv_kernel<<<g, b, 0, stream>>>(wo, wto);
        wconv_kernel<<<g, b, 0, stream>>>(w1, wt1);
        wconv_kernel<<<g, b, 0, stream>>>(w2, wt2);
    }

    // ---- embedding ----
    embed_kernel<<<cM, 256, 0, stream>>>(idx, emb, xf, xb);

    const dim3 gemmGrid(cD / 128, cM / 128);   // (6, 256)

    for (int l = 0; l < cNL; ++l) {
        const bf16* Wq = wtq + (size_t)l * DD;
        const bf16* Wk = wtk + (size_t)l * DD;
        const bf16* Wv = wtv + (size_t)l * DD;
        const bf16* Wo = wto + (size_t)l * DD;
        const bf16* W1 = wt1 + (size_t)l * DD;
        const bf16* W2 = wt2 + (size_t)l * DD;

        // q,k,v projections (f32 out)
        gemm_bf16_kernel<false,false,true,false><<<gemmGrid, 256, 0, stream>>>(
            xb, Wq, bq + (size_t)l * cD, nullptr, qb, nullptr);
        gemm_bf16_kernel<false,false,true,false><<<gemmGrid, 256, 0, stream>>>(
            xb, Wk, bk + (size_t)l * cD, nullptr, kb, nullptr);
        gemm_bf16_kernel<false,false,true,false><<<gemmGrid, 256, 0, stream>>>(
            xb, Wv, bv + (size_t)l * cD, nullptr, vb, nullptr);

        // RoPE + random features
        rope_phi_kernel<<<cM, 128, 0, stream>>>(qb, omega + (size_t)l * cHD * cNF, nullptr, Qf, 0);
        rope_phi_kernel<<<cM, 128, 0, stream>>>(kb, omega + (size_t)l * cHD * cNF, mask,    Kf, 1);

        // linear attention: KV / Ksum then combine
        kv_accum_kernel<<<cB * cH, 256, 0, stream>>>(Kf, vb, KV, Ks);
        attn_combine_kernel<<<dim3(cS / 32, cB * cH), 256, 0, stream>>>(Qf, KV, Ks, ab);

        // out proj + residual -> qb (tmp), then LN1 -> xf/xb
        gemm_bf16_kernel<false,true,true,false><<<gemmGrid, 256, 0, stream>>>(
            ab, Wo, bo + (size_t)l * cD, xf, qb, nullptr);
        ln_kernel<<<cM, 256, 0, stream>>>(qb, g1 + (size_t)l * cD, be1 + (size_t)l * cD, xf, xb);

        // FFN: gelu(x@w1+c1) (bf16 only) -> ab ; x + ab@w2+c2 -> qb ; LN2 -> xf/xb
        gemm_bf16_kernel<true,false,false,true><<<gemmGrid, 256, 0, stream>>>(
            xb, W1, c1 + (size_t)l * cD, nullptr, nullptr, ab);
        gemm_bf16_kernel<false,true,true,false><<<gemmGrid, 256, 0, stream>>>(
            ab, W2, c2 + (size_t)l * cD, xf, qb, nullptr);
        ln_kernel<<<cM, 256, 0, stream>>>(qb, g2 + (size_t)l * cD, be2 + (size_t)l * cD, xf, xb);
    }

    // final LayerNorm -> d_out (f32)
    ln_kernel<<<cM, 256, 0, stream>>>(xf, gf, bfp, (float*)d_out, nullptr);
}
